// ThreeLayerGraphSage_BN_75977971466899
// MI455X (gfx1250) — matrix operand order repaired and checked by reading, compiled, not run on hardware
//
#include <hip/hip_runtime.h>
#include <cstdint>

#define DFEAT 128
#define KDIM  256          // self(128) || neigh(128) concatenated K
#define BN_EPS 1e-5f

// ---------------- CDNA5 WMMA types ----------------
typedef __attribute__((ext_vector_type(16))) __bf16 v16bf;
typedef __attribute__((ext_vector_type(8)))  float  v8f;

union FragU {
    uint4 q[2];
    v16bf v;
};

// ---------------- bf16 helpers (RNE) ----------------
static __device__ __forceinline__ unsigned short f2bf(float f) {
    unsigned u = __builtin_bit_cast(unsigned, f);
    unsigned r = (u + 0x7FFFu + ((u >> 16) & 1u)) >> 16;
    return (unsigned short)r;
}
static __device__ __forceinline__ float bf2f(unsigned short h) {
    unsigned u = ((unsigned)h) << 16;
    return __builtin_bit_cast(float, u);
}
static __device__ __forceinline__ void split_bf(float x, unsigned short& hi, unsigned short& lo) {
    hi = f2bf(x);
    lo = f2bf(x - bf2f(hi));
}

// ---------------- utility kernels ----------------
__global__ void k_zero_f32(float* __restrict__ p, int n) {
    int i = blockIdx.x * blockDim.x + threadIdx.x;
    if (i < n) p[i] = 0.0f;
}
__global__ void k_zero_i32(int* __restrict__ p, int n) {
    int i = blockIdx.x * blockDim.x + threadIdx.x;
    if (i < n) p[i] = 0;
}

// ---------------- CSR build (once per launch, reused by all 3 layers) ----------------
__global__ void k_hist(const int* __restrict__ dst, int* __restrict__ cnt, int E) {
    int e = blockIdx.x * blockDim.x + threadIdx.x;
    if (e < E) atomicAdd(&cnt[dst[e]], 1);
}

// per-block exclusive scan (256 elements) via LDS Hillis-Steele; emits block sums
__global__ void k_scan_block(const int* __restrict__ cnt, int* __restrict__ excl,
                             int* __restrict__ bsum, int N) {
    __shared__ int sh[256];
    int i = blockIdx.x * 256 + threadIdx.x;
    int v = (i < N) ? cnt[i] : 0;
    sh[threadIdx.x] = v;
    __syncthreads();
    for (int ofs = 1; ofs < 256; ofs <<= 1) {
        int t = (threadIdx.x >= ofs) ? sh[threadIdx.x - ofs] : 0;
        __syncthreads();
        sh[threadIdx.x] += t;
        __syncthreads();
    }
    if (i < N) excl[i] = sh[threadIdx.x] - v;       // exclusive within block
    if (threadIdx.x == 255) bsum[blockIdx.x] = sh[255];
}

// tiny single-thread exclusive scan of the ~196 block sums
__global__ void k_scan_bsum(int* __restrict__ bsum, int nb) {
    if (blockIdx.x == 0 && threadIdx.x == 0) {
        int acc = 0;
        for (int b = 0; b < nb; ++b) { int t = bsum[b]; bsum[b] = acc; acc += t; }
    }
}

// add block offsets; also finalize rowptr[N] = E and init cursor = rowptr
__global__ void k_scan_add(int* __restrict__ rowptr, int* __restrict__ cursor,
                           const int* __restrict__ bsum, int N, int E) {
    int i = blockIdx.x * 256 + threadIdx.x;
    if (i < N) {
        int v = rowptr[i] + bsum[blockIdx.x];
        rowptr[i] = v;
        cursor[i] = v;
    }
    if (i == 0) rowptr[N] = E;
}

__global__ void k_fill(const int* __restrict__ src, const int* __restrict__ dst,
                       int* __restrict__ cursor, int* __restrict__ eidx, int E) {
    int e = blockIdx.x * blockDim.x + threadIdx.x;
    if (e < E) {
        int pos = atomicAdd(&cursor[dst[e]], 1);
        eidx[pos] = src[e];
    }
}

// ---------------- aggregation: pure gather, no atomics ----------------
// One wave per destination node; lane owns 4 features. Accumulates the neighbor mean in
// registers and writes bf16 hi/lo directly into the NEIGH half (cols 128..255) of [N][256].
__global__ void k_gather(const float* __restrict__ feat, const int* __restrict__ rowptr,
                         const int* __restrict__ eidx, unsigned short* __restrict__ fh,
                         unsigned short* __restrict__ fl, int N) {
    int node = (blockIdx.x * blockDim.x + threadIdx.x) >> 5;
    int lane = threadIdx.x & 31;
    if (node >= N) return;
    int j0 = rowptr[node], j1 = rowptr[node + 1];
    float ax = 0.f, ay = 0.f, az = 0.f, aw = 0.f;
    for (int j = j0; j < j1; ++j) {
        int s = eidx[j];
        float4 v = ((const float4*)(feat + (size_t)s * DFEAT))[lane];
        ax += v.x; ay += v.y; az += v.z; aw += v.w;
    }
    float inv = 1.0f / fmaxf((float)(j1 - j0), 1.0f);
    ax *= inv; ay *= inv; az *= inv; aw *= inv;

    unsigned short h0, l0, h1, l1, h2, l2, h3, l3;
    split_bf(ax, h0, l0); split_bf(ay, h1, l1);
    split_bf(az, h2, l2); split_bf(aw, h3, l3);

    size_t o = (size_t)node * KDIM + DFEAT + lane * 4;   // bf16 element offset (8B aligned)
    uint2 ph, pl;
    ph.x = (unsigned)h0 | ((unsigned)h1 << 16); ph.y = (unsigned)h2 | ((unsigned)h3 << 16);
    pl.x = (unsigned)l0 | ((unsigned)l1 << 16); pl.y = (unsigned)l2 | ((unsigned)l3 << 16);
    *(uint2*)(fh + o) = ph;
    *(uint2*)(fl + o) = pl;
}

// split f32 -> bf16 hi/lo into the SELF half (cols 0..127) of the [N][256] feature arrays
__global__ void k_split_self(const float* __restrict__ x, unsigned short* __restrict__ fh,
                             unsigned short* __restrict__ fl, int nf) {
    int i = blockIdx.x * blockDim.x + threadIdx.x;
    if (i >= nf) return;
    int row = i >> 7, c = i & (DFEAT - 1);
    size_t o = (size_t)row * KDIM + c;
    unsigned short h, l;
    split_bf(x[i], h, l);
    fh[o] = h; fl[o] = l;
}

// Pack combined [Ws; Wn] (256x128) into wave32 WMMA B-fragment layout, hi/lo bf16.
// lane holds column n = nt*16 + (lane&15); element i -> K = kt*32 + (i&7) + 8*(lane>>4) + 16*(i>>3)
__global__ void k_prepw(const float* __restrict__ Ws, const float* __restrict__ Wn,
                        unsigned short* __restrict__ ph, unsigned short* __restrict__ pl) {
    int idx = blockIdx.x * blockDim.x + threadIdx.x;   // 8*8*32*16 = 32768 entries
    if (idx >= 8 * 8 * 32 * 16) return;
    int i    = idx & 15;
    int lane = (idx >> 4) & 31;
    int nt   = (idx >> 9) & 7;
    int kt   = (idx >> 12) & 7;
    int kh   = lane >> 4;
    int n    = nt * 16 + (lane & 15);
    int k    = kt * 32 + (i & 7) + 8 * kh + 16 * (i >> 3); // 0..255
    float w = (k < DFEAT) ? Ws[k * DFEAT + n] : Wn[(k - DFEAT) * DFEAT + n];
    unsigned short h, l;
    split_bf(w, h, l);
    ph[idx] = h; pl[idx] = l;
}

// ---------------- WMMA GEMM (unchanged: spill-free codegen confirmed) ----------------
// out[N x 128] = F[N x 256] @ Wcat[256 x 128] + bias, split-bf16 (hi*hi + hi*lo + lo*hi).
// Each wave owns a 16x64 output tile (4 x v8f accumulators); two waves cover a row tile.
__global__ void k_gemm(const unsigned short* __restrict__ fh, const unsigned short* __restrict__ fl,
                       const unsigned short* __restrict__ wh, const unsigned short* __restrict__ wl,
                       const float* __restrict__ bias, float* __restrict__ out, int nwaves) {
    const int wid  = (blockIdx.x * (blockDim.x >> 5)) + (threadIdx.x >> 5);
    const int lane = threadIdx.x & 31;
    if (wid >= nwaves) return;

    const int r  = wid >> 1;        // row tile
    const int nh = wid & 1;         // n half: n-tiles nh*4 .. nh*4+3
    const int m  = lane & 15;
    const int kh = lane >> 4;
    const size_t row = (size_t)r * 16 + m;

    v8f acc[4];
#pragma unroll
    for (int i = 0; i < 4; ++i) acc[i] = v8f{};

    const uint4* wh4 = (const uint4*)wh;
    const uint4* wl4 = (const uint4*)wl;

    for (int kt = 0; kt < 8; ++kt) {
        const size_t aoff = row * KDIM + (size_t)kt * 32 + 8 * kh;
        FragU ah, al;
        ah.q[0] = *(const uint4*)(fh + aoff); ah.q[1] = *(const uint4*)(fh + aoff + 16);
        al.q[0] = *(const uint4*)(fl + aoff); al.q[1] = *(const uint4*)(fl + aoff + 16);

#pragma unroll
        for (int t = 0; t < 4; ++t) {
            const int nt = nh * 4 + t;
            const int bidx = (((kt * 8 + nt) * 32) + lane) * 2;  // uint4 index (32B frags)
            FragU bh, bl;
            bh.q[0] = wh4[bidx]; bh.q[1] = wh4[bidx + 1];
            bl.q[0] = wl4[bidx]; bl.q[1] = wl4[bidx + 1];

            acc[t] = __builtin_amdgcn_wmma_f32_16x16x32_bf16(false, ah.v, false, bh.v, (short)0, acc[t], false, false);
            acc[t] = __builtin_amdgcn_wmma_f32_16x16x32_bf16(false, ah.v, false, bl.v, (short)0, acc[t], false, false);
            acc[t] = __builtin_amdgcn_wmma_f32_16x16x32_bf16(false, al.v, false, bh.v, (short)0, acc[t], false, false);
        }
    }

    const int rbase = r * 16 + 8 * kh;
#pragma unroll
    for (int t = 0; t < 4; ++t) {
        const int n = (nh * 4 + t) * 16 + m;
        const float bv = bias[n];
#pragma unroll
        for (int j = 0; j < 8; ++j) {
            out[(size_t)(rbase + j) * DFEAT + n] = acc[t][j] + bv;
        }
    }
}

// ---------------- BatchNorm ----------------
__global__ void k_bnstats(const float* __restrict__ h, float* __restrict__ sums,
                          float* __restrict__ ssq, int N) {
    int c = threadIdx.x;                              // 0..127 (one column per thread)
    int rpb = (N + gridDim.x - 1) / gridDim.x;
    int r0 = blockIdx.x * rpb;
    int r1 = r0 + rpb; if (r1 > N) r1 = N;
    float s = 0.0f, q = 0.0f;
    for (int r = r0; r < r1; ++r) {
        float v = h[(size_t)r * DFEAT + c];
        s += v; q += v * v;
    }
    atomicAdd(&sums[c], s);
    atomicAdd(&ssq[c], q);
}

__global__ void k_bnnorm(float* __restrict__ h, const float* __restrict__ sums,
                         const float* __restrict__ ssq, const float* __restrict__ gamma,
                         const float* __restrict__ beta, unsigned short* __restrict__ fh,
                         unsigned short* __restrict__ fl, int nf, int N) {
    int i = blockIdx.x * blockDim.x + threadIdx.x;
    if (i >= nf) return;
    int row = i >> 7, c = i & (DFEAT - 1);
    float invN = 1.0f / (float)N;
    float mu  = sums[c] * invN;
    float var = ssq[c] * invN - mu * mu;
    float y = (h[i] - mu) * rsqrtf(var + BN_EPS) * gamma[c] + beta[c];
    y = fmaxf(y, 0.0f);
    h[i] = y;
    size_t o = (size_t)row * KDIM + c;
    unsigned short hi, lo;
    split_bf(y, hi, lo);
    fh[o] = hi; fl[o] = lo;
}

// ---------------- host launcher ----------------
extern "C" void kernel_launch(void* const* d_in, const int* in_sizes, int n_in,
                              void* d_out, int out_size, void* d_ws, size_t ws_size,
                              hipStream_t stream) {
    const float* x   = (const float*)d_in[0];
    const int*   src = (const int*)d_in[1];
    const int*   dst = (const int*)d_in[2];
    const float* Ws1 = (const float*)d_in[3];
    const float* Wn1 = (const float*)d_in[4];
    const float* b1  = (const float*)d_in[5];
    const float* g1  = (const float*)d_in[6];
    const float* be1 = (const float*)d_in[7];
    const float* Ws2 = (const float*)d_in[8];
    const float* Wn2 = (const float*)d_in[9];
    const float* b2  = (const float*)d_in[10];
    const float* g2  = (const float*)d_in[11];
    const float* be2 = (const float*)d_in[12];
    const float* Ws3 = (const float*)d_in[13];
    const float* Wn3 = (const float*)d_in[14];
    const float* b3  = (const float*)d_in[15];

    const int N  = in_sizes[0] / DFEAT;   // 50000
    const int E  = in_sizes[1];           // 800000
    const int NF = N * DFEAT;
    const int tiles = N / 16;             // 3125 (exact)
    const int nb = (N + 255) / 256;       // scan blocks

    // ---- carve workspace ----
    char* ws = (char*)d_ws;
    size_t off = 0;
    auto carve = [&](size_t bytes) -> void* {
        void* p = ws + off;
        off = (off + bytes + 255) & ~(size_t)255;
        return p;
    };
    int* degc   = (int*)carve((size_t)N * 4);
    int* rowptr = (int*)carve((size_t)(N + 1) * 4);
    int* cursor = (int*)carve((size_t)N * 4);
    int* bsum   = (int*)carve((size_t)nb * 4);
    int* eidx   = (int*)carve((size_t)E * 4);
    float* h    = (float*)carve((size_t)NF * 4);
    unsigned short* fhi = (unsigned short*)carve((size_t)N * KDIM * 2);  // [N][256] bf16 hi
    unsigned short* flo = (unsigned short*)carve((size_t)N * KDIM * 2);  // [N][256] bf16 lo
    const size_t WPK = 8 * 8 * 32 * 16;   // 32768 packed bf16 per combined weight
    unsigned short* pw[6];                // 3 layers x (hi, lo)
    for (int i = 0; i < 6; ++i) pw[i] = (unsigned short*)carve(WPK * 2);
    float* bn = (float*)carve(256 * 4);
    float* bnsum = bn;
    float* bnssq = bn + 128;
    (void)ws_size; (void)n_in; (void)out_size;

    const int TB = 256;
    auto zgrid = [&](int n) { return (n + TB - 1) / TB; };

    // ---- weight prep: pack [Ws; Wn] hi/lo into WMMA B-frag layout ----
    const float* WsA[3] = { Ws1, Ws2, Ws3 };
    const float* WnA[3] = { Wn1, Wn2, Wn3 };
    for (int i = 0; i < 3; ++i)
        k_prepw<<<(int)((WPK + TB - 1) / TB), TB, 0, stream>>>(WsA[i], WnA[i], pw[2 * i], pw[2 * i + 1]);

    // ---- CSR build (once; reused by all 3 layers) ----
    k_zero_i32<<<zgrid(N), TB, 0, stream>>>(degc, N);
    k_hist<<<zgrid(E), TB, 0, stream>>>(dst, degc, E);
    k_scan_block<<<nb, 256, 0, stream>>>(degc, rowptr, bsum, N);
    k_scan_bsum<<<1, 1, 0, stream>>>(bsum, nb);
    k_scan_add<<<nb, 256, 0, stream>>>(rowptr, cursor, bsum, N, E);
    k_fill<<<zgrid(E), TB, 0, stream>>>(src, dst, cursor, eidx, E);

    // ---- self features for layer 1 ----
    k_split_self<<<zgrid(NF), TB, 0, stream>>>(x, fhi, flo, NF);

    const int gatherBlocks = (N * 32 + TB - 1) / TB;    // one wave per node
    const int nwaves = tiles * 2;                       // 16x64 tile per wave
    const int gemmBlocks = (nwaves + 7) / 8;            // 8 waves per 256-thread block

    auto layer = [&](const float* featF32, const unsigned short* wh_, const unsigned short* wl_,
                     const float* bias, float* outp) {
        k_gather<<<gatherBlocks, TB, 0, stream>>>(featF32, rowptr, eidx, fhi, flo, N);
        k_gemm<<<gemmBlocks, TB, 0, stream>>>(fhi, flo, wh_, wl_, bias, outp, nwaves);
    };
    auto bn_relu = [&](const float* gamma, const float* beta) {
        k_zero_f32<<<1, TB, 0, stream>>>(bn, 256);
        k_bnstats<<<200, 128, 0, stream>>>(h, bnsum, bnssq, N);
        k_bnnorm<<<zgrid(NF), TB, 0, stream>>>(h, bnsum, bnssq, gamma, beta, fhi, flo, NF, N);
    };

    // Layer 1: gather(x); h = [x||n] @ [Ws1;Wn1] + b1 ; BN1+ReLU (refreshes self half of fhi/flo)
    layer(x, pw[0], pw[1], b1, h);
    bn_relu(g1, be1);
    // Layer 2
    layer(h, pw[2], pw[3], b2, h);
    bn_relu(g2, be2);
    // Layer 3 -> d_out (no BN)
    layer(h, pw[4], pw[5], b3, (float*)d_out);
}
